// RoI_5772436046020
// MI455X (gfx1250) — matrix operand order repaired and checked by reading, not compiled, not run on hardware
//
#include <hip/hip_runtime.h>
#include <hip/hip_bf16.h>

typedef _Float16 v16h __attribute__((ext_vector_type(16)));
typedef _Float16 v8h  __attribute__((ext_vector_type(8)));
typedef float    v8f  __attribute__((ext_vector_type(8)));
typedef int      v4i  __attribute__((ext_vector_type(4)));

typedef __attribute__((address_space(1))) v4i* gas1_v4i;
typedef __attribute__((address_space(3))) v4i* las3_v4i;

#define IMG_SZ 800.0f
#define FSCALE 0.25f
#define ROUT 7
#define NUMC 7
#define SCORE_TH 0.05f
#define NMS_TH 0.5f
#define DETS 100
#define NEGV -1.0e30f
#define BBOX_CLIP 4.135166556742356f  // log(1000/16)

#define BN 2
#define PN 300
#define CN 512
#define HN 200
#define WN 200
#define NROI (BN * PN)            // 600
#define D1 (CN * ROUT * ROUT)     // 25088
#define HID 1024
#define MPAD 640                  // 600 -> 40 full M-tiles (5 blocks x 8 waves)
#define MTILES (MPAD / 16)        // 40
#define CAND (PN * (NUMC - 1))    // 1800 per image
#define KCHUNK 128                // K staged per LDS buffer
#define NSTRIP 64                 // columns per block
#define KSPLIT 4                  // split-K factor

// ---------------------------------------------------------------------------
// CDNA5 async global->LDS helpers (ASYNCcnt-tracked), builtin or asm fallback
// ---------------------------------------------------------------------------
__device__ __forceinline__ void async_load_b128(const void* g, void* l) {
#if __has_builtin(__builtin_amdgcn_global_load_async_to_lds_b128)
    __builtin_amdgcn_global_load_async_to_lds_b128((gas1_v4i)g, (las3_v4i)l, 0, 0);
#else
    unsigned lo = (unsigned)(uintptr_t)l;
    unsigned long long ga = (unsigned long long)(uintptr_t)g;
    asm volatile("global_load_async_to_lds_b128 %0, %1, off"
                 :: "v"(lo), "v"(ga) : "memory");
#endif
}

template <int N>
__device__ __forceinline__ void wait_asynccnt() {
#if __has_builtin(__builtin_amdgcn_s_wait_asynccnt)
    __builtin_amdgcn_s_wait_asynccnt((unsigned short)N);
#else
    asm volatile("s_wait_asynccnt %0" :: "i"(N) : "memory");
#endif
}

// ---------------------------------------------------------------------------
// 1) Tiled transpose + f32->f16 convert: W (K x Nc f32) -> Wt (Nc x K f16)
// ---------------------------------------------------------------------------
__global__ __launch_bounds__(256)
void transpose_to_f16(const float* __restrict__ W, _Float16* __restrict__ Wt,
                      int K, int Nc) {
    __shared__ float tile[32][33];
    int tx = threadIdx.x;          // 0..31
    int ty = threadIdx.y;          // 0..7
    int colBase = blockIdx.x * 32; // over Nc
    int rowBase = blockIdx.y * 32; // over K
#pragma unroll
    for (int i = 0; i < 4; ++i) {
        int r = rowBase + ty + i * 8;
        tile[ty + i * 8][tx] = W[(size_t)r * Nc + colBase + tx];
    }
    __syncthreads();
#pragma unroll
    for (int i = 0; i < 4; ++i) {
        int n = colBase + ty + i * 8;
        Wt[(size_t)n * K + rowBase + tx] = (_Float16)tile[tx][ty + i * 8];
    }
}

// ---------------------------------------------------------------------------
// 2) ROI align -> x_f16 (MPAD x D1), one thread per output element.
// ---------------------------------------------------------------------------
__global__ __launch_bounds__(256)
void roi_align_kernel(const float* __restrict__ features,
                      const float* __restrict__ proposals,
                      _Float16* __restrict__ xout) {
    size_t idx = (size_t)blockIdx.x * blockDim.x + threadIdx.x;
    if (idx >= (size_t)NROI * D1) return;
    int n = (int)(idx / D1);
    int d = (int)(idx % D1);
    int c = d / (ROUT * ROUT);
    int rr = d % (ROUT * ROUT);
    int oy = rr / ROUT;
    int ox = rr % ROUT;
    int b = n / PN;

    float x1 = proposals[n * 4 + 0] * FSCALE;
    float y1 = proposals[n * 4 + 1] * FSCALE;
    float x2 = proposals[n * 4 + 2] * FSCALE;
    float y2 = proposals[n * 4 + 3] * FSCALE;
    float rw = fmaxf(x2 - x1, 1.0f);
    float rh = fmaxf(y2 - y1, 1.0f);
    float bin_w = rw * (1.0f / ROUT);
    float bin_h = rh * (1.0f / ROUT);

    const float* fbase = features + ((size_t)(b * CN + c)) * (HN * WN);
    float sum = 0.0f;
#pragma unroll
    for (int sy = 0; sy < 2; ++sy) {
#pragma unroll
        for (int sx = 0; sx < 2; ++sx) {
            float gy = ((float)(oy * 2 + sy) + 0.5f) * 0.5f;
            float gx = ((float)(ox * 2 + sx) + 0.5f) * 0.5f;
            float yy = y1 + gy * bin_h;
            float xx = x1 + gx * bin_w;
            bool valid = (yy > -1.0f) && (yy < (float)HN) &&
                         (xx > -1.0f) && (xx < (float)WN);
            float yc = fminf(fmaxf(yy, 0.0f), (float)(HN - 1));
            float xc = fminf(fmaxf(xx, 0.0f), (float)(WN - 1));
            float y0f = floorf(yc), x0f = floorf(xc);
            int y0 = (int)y0f, x0 = (int)x0f;
            int y1i = min(y0 + 1, HN - 1);
            int x1i = min(x0 + 1, WN - 1);
            float ly = yc - y0f, lx = xc - x0f;
            float v00 = fbase[y0 * WN + x0];
            float v01 = fbase[y0 * WN + x1i];
            float v10 = fbase[y1i * WN + x0];
            float v11 = fbase[y1i * WN + x1i];
            float v = (1.0f - ly) * ((1.0f - lx) * v00 + lx * v01) +
                      ly * ((1.0f - lx) * v10 + lx * v11);
            sum += valid ? v : 0.0f;
        }
    }
    xout[idx] = (_Float16)(sum * 0.25f);
}

// ---------------------------------------------------------------------------
// 3) Split-K WMMA GEMM producing f32 partials.
//    A (MPAD x K f16 row-major), Bt (N x K f16, pre-transposed weights).
//    Block = 8 waves = 8 M-tiles x one 64-col strip, blockIdx.z = K split.
//    B panel staged to LDS via async global->LDS b128 (double-buffered,
//    KCHUNK=128 per stage) and shared by all 8 waves via ds_load.
// ---------------------------------------------------------------------------
__global__ __launch_bounds__(256)
void wmma_gemm_partial(const _Float16* __restrict__ A,
                       const _Float16* __restrict__ Bt,
                       float* __restrict__ Pout,   // [KSPLIT][MPAD][N]
                       int K, int N, int kLen) {
    __shared__ __align__(32) _Float16 sB[2][NSTRIP * KCHUNK];  // 2 x 16KB

    int tid = threadIdx.x;
    int wave = tid >> 5;
    int lane = tid & 31;
    int mBase = (blockIdx.y * 8 + wave) * 16;
    int nBase = blockIdx.x * NSTRIP;
    int kBase = blockIdx.z * kLen;
    int l16 = lane & 15;
    int hs = lane >> 4;

    const _Float16* aRow = A + (size_t)(mBase + l16) * K + kBase + hs * 16;
    const char* bt8 = (const char*)Bt;

    v8f acc[4];
#pragma unroll
    for (int t = 0; t < 4; ++t) acc[t] = (v8f){0, 0, 0, 0, 0, 0, 0, 0};

    // stage layout: col-major panels, 64 cols x KCHUNK halves (256B per col)
    auto issue = [&](int chunk, int buf) {
        size_t kb = (size_t)(kBase + chunk * KCHUNK) * 2;
#pragma unroll
        for (int i = 0; i < 4; ++i) {
            int o = (i * 256 + tid) * 16;   // byte offset in 16KB stage
            int col = o >> 8;               // 256 bytes per column
            int inCol = o & 255;
            const char* g = bt8 + (size_t)(nBase + col) * K * 2 + kb + inCol;
            async_load_b128(g, (char*)&sB[buf][0] + o);
        }
    };

    int numChunks = kLen / KCHUNK;
    issue(0, 0);
    for (int c = 0; c < numChunks; ++c) {
        int buf = c & 1;
        if (c + 1 < numChunks) {
            issue(c + 1, buf ^ 1);
            wait_asynccnt<4>();   // oldest 4 (current chunk) complete
        } else {
            wait_asynccnt<0>();
        }
        __syncthreads();

        const _Float16* aC = aRow + c * KCHUNK;
#pragma unroll
        for (int ks = 0; ks < KCHUNK / 32; ++ks) {
            v8h alo = *(const v8h*)(aC + ks * 32);
            v8h ahi = *(const v8h*)(aC + ks * 32 + 8);
            v16h a = __builtin_shufflevector(alo, ahi, 0, 1, 2, 3, 4, 5, 6, 7,
                                             8, 9, 10, 11, 12, 13, 14, 15);
            // hoist all 4 B fragments so the 8 ds_load_b128 issue as one
            // clause and waits can be partial instead of dscnt==0 per tile
            v16h bf[4];
#pragma unroll
            for (int t = 0; t < 4; ++t) {
                const _Float16* bp =
                    &sB[buf][(t * 16 + l16) * KCHUNK + ks * 32 + hs * 16];
                bf[t] = *(const v16h*)bp;
            }
#pragma unroll
            for (int t = 0; t < 4; ++t) {
                acc[t] = __builtin_amdgcn_wmma_f32_16x16x32_f16(
                    false, a, false, bf[t], (short)0, acc[t], false, false);
            }
        }
        __syncthreads();   // all waves done reading before buffer is reused
    }

    float* P = Pout + (size_t)blockIdx.z * MPAD * N;
#pragma unroll
    for (int t = 0; t < 4; ++t) {
        int col = nBase + t * 16 + l16;
#pragma unroll
        for (int v = 0; v < 8; ++v) {
            int row = mBase + v + hs * 8;
            P[(size_t)row * N + col] = acc[t][v];
        }
    }
}

// ---------------------------------------------------------------------------
// 3b) Combine split-K partials + bias + ReLU -> f16 or f32.
// ---------------------------------------------------------------------------
template <bool OUT_F16>
__global__ __launch_bounds__(256)
void combine_bias_relu(const float* __restrict__ P, const float* __restrict__ bias,
                       void* __restrict__ Out, int N) {
    size_t idx = (size_t)blockIdx.x * blockDim.x + threadIdx.x;
    if (idx >= (size_t)MPAD * N) return;
    int col = (int)(idx % N);
    float sum = 0.0f;
#pragma unroll
    for (int z = 0; z < KSPLIT; ++z) sum += P[(size_t)z * MPAD * N + idx];
    float val = fmaxf(sum + bias[col], 0.0f);
    if (OUT_F16)
        ((_Float16*)Out)[idx] = (_Float16)val;
    else
        ((float*)Out)[idx] = val;
}

// ---------------------------------------------------------------------------
// 4) Head: logits (600x7) and deltas (600x28) from h2 (600x1024 f32).
// ---------------------------------------------------------------------------
__global__ __launch_bounds__(256)
void head_kernel(const float* __restrict__ h2,
                 const float* __restrict__ wc, const float* __restrict__ bc,
                 const float* __restrict__ wb, const float* __restrict__ bb,
                 float* __restrict__ logits, float* __restrict__ deltas) {
    int t = blockIdx.x * blockDim.x + threadIdx.x;
    if (t >= NROI * (NUMC + NUMC * 4)) return;
    int n = t / (NUMC + NUMC * 4);
    int j = t % (NUMC + NUMC * 4);
    const float* h = h2 + (size_t)n * HID;
    float sum;
    if (j < NUMC) {
        sum = bc[j];
        for (int k = 0; k < HID; ++k) sum += h[k] * wc[k * NUMC + j];
        logits[n * NUMC + j] = sum;
    } else {
        int jj = j - NUMC;
        sum = bb[jj];
        for (int k = 0; k < HID; ++k) sum += h[k] * wb[k * NUMC * 4 + jj];
        deltas[n * NUMC * 4 + jj] = sum;
    }
}

// ---------------------------------------------------------------------------
// 5) Softmax + box decode -> per-image candidate lists (box, initial score).
// ---------------------------------------------------------------------------
__global__ __launch_bounds__(256)
void decode_kernel(const float* __restrict__ proposals,
                   const float* __restrict__ logits,
                   const float* __restrict__ deltas,
                   float* __restrict__ candBox, float* __restrict__ candSw) {
    int n = blockIdx.x * blockDim.x + threadIdx.x;
    if (n >= NROI) return;
    int b = n / PN, p = n % PN;

    float lg[NUMC], mx = -3.0e38f;
#pragma unroll
    for (int k = 0; k < NUMC; ++k) { lg[k] = logits[n * NUMC + k]; mx = fmaxf(mx, lg[k]); }
    float se = 0.0f;
#pragma unroll
    for (int k = 0; k < NUMC; ++k) { lg[k] = __expf(lg[k] - mx); se += lg[k]; }
    float inv = 1.0f / se;

    float bx1 = proposals[n * 4 + 0];
    float by1 = proposals[n * 4 + 1];
    float bx2 = proposals[n * 4 + 2];
    float by2 = proposals[n * 4 + 3];
    float w = bx2 - bx1, h = by2 - by1;
    float cx = bx1 + 0.5f * w, cy = by1 + 0.5f * h;

    for (int k = 1; k < NUMC; ++k) {
        const float* dl = deltas + n * NUMC * 4 + k * 4;
        float dx = dl[0] * 0.1f;
        float dy = dl[1] * 0.1f;
        float dw = fminf(dl[2] * 0.2f, BBOX_CLIP);
        float dh = fminf(dl[3] * 0.2f, BBOX_CLIP);
        float pcx = dx * w + cx, pcy = dy * h + cy;
        float pw = __expf(dw) * w, ph = __expf(dh) * h;
        float o0 = fminf(fmaxf(pcx - 0.5f * pw, 0.0f), IMG_SZ);
        float o1 = fminf(fmaxf(pcy - 0.5f * ph, 0.0f), IMG_SZ);
        float o2 = fminf(fmaxf(pcx + 0.5f * pw, 0.0f), IMG_SZ);
        float o3 = fminf(fmaxf(pcy + 0.5f * ph, 0.0f), IMG_SZ);
        float pr = lg[k] * inv;
        bool valid = (pr > SCORE_TH) && ((o2 - o0) >= 0.01f) && ((o3 - o1) >= 0.01f);
        int m = b * CAND + p * (NUMC - 1) + (k - 1);
        candBox[m * 4 + 0] = o0;
        candBox[m * 4 + 1] = o1;
        candBox[m * 4 + 2] = o2;
        candBox[m * 4 + 3] = o3;
        candSw[m] = valid ? pr : NEGV;
    }
}

// ---------------------------------------------------------------------------
// 6) Per-image sequential NMS in LDS; one block per image.
//    Output layout: boxes[B*100*4] | scores[B*100] | labels[B*100] (floats).
// ---------------------------------------------------------------------------
__global__ __launch_bounds__(256)
void nms_kernel(const float* __restrict__ candBox,
                const float* __restrict__ candSw,
                float* __restrict__ out) {
    __shared__ float sOb[CAND * 4];
    __shared__ float sArea[CAND];
    __shared__ float sSw[CAND];
    __shared__ float rVal[256];
    __shared__ int   rIdx[256];

    int b = blockIdx.x;
    int tid = threadIdx.x;
    const float* cb = candBox + (size_t)b * CAND * 4;
    const float* cs = candSw + (size_t)b * CAND;

    for (int m = tid; m < CAND; m += 256) {
        float off = (float)((m % (NUMC - 1)) + 1) * (IMG_SZ + 2.0f);
        float x1 = cb[m * 4 + 0] + off;
        float y1 = cb[m * 4 + 1] + off;
        float x2 = cb[m * 4 + 2] + off;
        float y2 = cb[m * 4 + 3] + off;
        sOb[m * 4 + 0] = x1; sOb[m * 4 + 1] = y1;
        sOb[m * 4 + 2] = x2; sOb[m * 4 + 3] = y2;
        sArea[m] = (x2 - x1) * (y2 - y1);
        sSw[m] = cs[m];
    }
    __syncthreads();

    for (int it = 0; it < DETS; ++it) {
        // argmax with first-index tie-break (matches jnp.argmax)
        float bv = -3.0e38f; int bi = CAND;
        for (int m = tid; m < CAND; m += 256) {
            float v = sSw[m];
            if (v > bv || (v == bv && m < bi)) { bv = v; bi = m; }
        }
        rVal[tid] = bv; rIdx[tid] = bi;
        __syncthreads();
        for (int s = 128; s > 0; s >>= 1) {
            if (tid < s) {
                float ov = rVal[tid + s]; int oi = rIdx[tid + s];
                if (ov > rVal[tid] || (ov == rVal[tid] && oi < rIdx[tid])) {
                    rVal[tid] = ov; rIdx[tid] = oi;
                }
            }
            __syncthreads();
        }
        int j = rIdx[0];
        float s = rVal[0];
        float offj = (float)((j % (NUMC - 1)) + 1) * (IMG_SZ + 2.0f);
        float jx1 = sOb[j * 4 + 0], jy1 = sOb[j * 4 + 1];
        float jx2 = sOb[j * 4 + 2], jy2 = sOb[j * 4 + 3];
        float areaJ = sArea[j];

        if (tid == 0) {
            bool keep = s > NEGV * 0.5f;
            int lab = (j % (NUMC - 1)) + 1;
            float* ob = out + ((size_t)b * DETS + it) * 4;
            ob[0] = keep ? (jx1 - offj) : 0.0f;
            ob[1] = keep ? (jy1 - offj) : 0.0f;
            ob[2] = keep ? (jx2 - offj) : 0.0f;
            ob[3] = keep ? (jy2 - offj) : 0.0f;
            out[BN * DETS * 4 + b * DETS + it] = keep ? s : 0.0f;
            out[BN * DETS * 5 + b * DETS + it] = keep ? (float)lab : 0.0f;
        }
        // suppress
        for (int m = tid; m < CAND; m += 256) {
            float xx1 = fmaxf(jx1, sOb[m * 4 + 0]);
            float yy1 = fmaxf(jy1, sOb[m * 4 + 1]);
            float xx2 = fminf(jx2, sOb[m * 4 + 2]);
            float yy2 = fminf(jy2, sOb[m * 4 + 3]);
            float inter = fmaxf(xx2 - xx1, 0.0f) * fmaxf(yy2 - yy1, 0.0f);
            float iou = inter / (areaJ + sArea[m] - inter + 1e-9f);
            if (iou > NMS_TH || m == j) sSw[m] = NEGV;
        }
        __syncthreads();
    }
}

// ---------------------------------------------------------------------------
extern "C" void kernel_launch(void* const* d_in, const int* in_sizes, int n_in,
                              void* d_out, int out_size, void* d_ws, size_t ws_size,
                              hipStream_t stream) {
    const float* features  = (const float*)d_in[0];
    const float* proposals = (const float*)d_in[1];
    const float* w1 = (const float*)d_in[2];
    const float* b1 = (const float*)d_in[3];
    const float* w2 = (const float*)d_in[4];
    const float* b2 = (const float*)d_in[5];
    const float* wc = (const float*)d_in[6];
    const float* bc = (const float*)d_in[7];
    const float* wb = (const float*)d_in[8];
    const float* bb = (const float*)d_in[9];
    float* out = (float*)d_out;

    char* ws = (char*)d_ws;
    size_t off = 0;
    auto alloc = [&](size_t bytes) {
        char* p = ws + off;
        off += (bytes + 255) & ~(size_t)255;
        return p;
    };
    _Float16* x_f16  = (_Float16*)alloc((size_t)MPAD * D1 * 2);
    _Float16* w1t    = (_Float16*)alloc((size_t)HID * D1 * 2);
    _Float16* w2t    = (_Float16*)alloc((size_t)HID * HID * 2);
    _Float16* h1     = (_Float16*)alloc((size_t)MPAD * HID * 2);
    float*    h2     = (float*)   alloc((size_t)MPAD * HID * 4);
    float*    part   = (float*)   alloc((size_t)KSPLIT * MPAD * HID * 4);
    float*    logits = (float*)   alloc((size_t)NROI * NUMC * 4);
    float*    deltas = (float*)   alloc((size_t)NROI * NUMC * 4 * 4);
    float*    cbox   = (float*)   alloc((size_t)BN * CAND * 4 * 4);
    float*    csw    = (float*)   alloc((size_t)BN * CAND * 4);

    // weight transposes (f32 -> f16, K x N -> N x K)
    transpose_to_f16<<<dim3(HID / 32, D1 / 32), dim3(32, 8), 0, stream>>>(w1, w1t, D1, HID);
    transpose_to_f16<<<dim3(HID / 32, HID / 32), dim3(32, 8), 0, stream>>>(w2, w2t, HID, HID);

    // ROI align -> f16 activations
    {
        size_t total = (size_t)NROI * D1;
        int blocks = (int)((total + 255) / 256);
        roi_align_kernel<<<blocks, 256, 0, stream>>>(features, proposals, x_f16);
    }

    int combBlocks = (int)(((size_t)MPAD * HID + 255) / 256);

    // GEMM1: x(640x25088) @ w1t -> split-K partials -> bias+relu -> h1 (f16)
    wmma_gemm_partial<<<dim3(HID / NSTRIP, MTILES / 8, KSPLIT), 256, 0, stream>>>(
        x_f16, w1t, part, D1, HID, D1 / KSPLIT);
    combine_bias_relu<true><<<combBlocks, 256, 0, stream>>>(part, b1, (void*)h1, HID);

    // GEMM2: h1(640x1024) @ w2t -> split-K partials -> bias+relu -> h2 (f32)
    wmma_gemm_partial<<<dim3(HID / NSTRIP, MTILES / 8, KSPLIT), 256, 0, stream>>>(
        h1, w2t, part, HID, HID, HID / KSPLIT);
    combine_bias_relu<false><<<combBlocks, 256, 0, stream>>>(part, b2, (void*)h2, HID);

    // classifier / regressor heads
    {
        int total = NROI * (NUMC + NUMC * 4);
        head_kernel<<<(total + 255) / 256, 256, 0, stream>>>(h2, wc, bc, wb, bb, logits, deltas);
    }

    decode_kernel<<<(NROI + 255) / 256, 256, 0, stream>>>(proposals, logits, deltas, cbox, csw);

    nms_kernel<<<BN, 256, 0, stream>>>(cbox, csw, out);
}
